// ViSNetEncoderMSE_43593918054752
// MI455X (gfx1250) — matrix operand order repaired and checked by reading, not compiled, not run on hardware
//
#include <hip/hip_runtime.h>
#include <hip/hip_bf16.h>
#include <math.h>

// ---------------------------------------------------------------------------
// ViSNet encoder forward for MI455X (gfx1250), round 3.
//  - All dense linears via WMMA f32_16x16x32_f16 (f16 operands, f32 accum).
//  - Weight strip (64 x K) staged in LDS once per block, read via ds_load_b128;
//    each wave computes a 16x64 strip -> 4 WMMAs per A-tile load.
//  - B lane layout same as A: lanes 0-15 hold K={0..7,16..23} of their column,
//    lanes 16-31 hold K={8..15,24..31} (the documented split layout).
//  - SiLU via v_rcp_f32 (no IEEE div sequence).
// Graph structure: edge e has dst = e/16, so segment_sum is a contiguous
// 16-edge reduction per node (no atomics).
// ---------------------------------------------------------------------------

#define NN 4096
#define EE 65536
#define HH 128
#define DEG 16
#define NUM_RBF 32
#define CUTOFF 5.0f
#define PI_F 3.14159265358979323846f

typedef __attribute__((ext_vector_type(16))) _Float16 v16h;
typedef __attribute__((ext_vector_type(8)))  _Float16 v8h;
typedef __attribute__((ext_vector_type(4)))  _Float16 v4h;
typedef __attribute__((ext_vector_type(8)))  float    v8f;
typedef __attribute__((ext_vector_type(4)))  float    v4f;

__device__ __forceinline__ float silu_f(float x) {
    return x * __builtin_amdgcn_rcpf(1.0f + __expf(-x));
}

// ---------------------------------------------------------------------------
// f32 -> f16 pack (4 elements/thread). n must be a multiple of 4.
// ---------------------------------------------------------------------------
__global__ void cvt_f16_k(const float* __restrict__ x, _Float16* __restrict__ y,
                          int n)
{
    int i = (blockIdx.x * blockDim.x + threadIdx.x) * 4;
    if (i >= n) return;
    v4f v = *(const v4f*)(x + i);
    v4h h;
    h[0] = (_Float16)v[0]; h[1] = (_Float16)v[1];
    h[2] = (_Float16)v[2]; h[3] = (_Float16)v[3];
    *(v4h*)(y + i) = h;
}

// ---------------------------------------------------------------------------
// Main NT GEMM with LDS-staged weights.
// C[M x Nout] = act(A[M x K] @ W[Nout x K]^T + bias)
// grid = (M/128, Nout/64); block = 256 (8 waves); each wave: 16 rows x 64 cols.
// Requires: K % 32 == 0, K <= 256, M % 128 == 0, Nout % 64 == 0.
// act: 0 = identity, 1 = SiLU.
// ---------------------------------------------------------------------------
__global__ __launch_bounds__(256) void gemm_h16_lds(
    const _Float16* __restrict__ A, const _Float16* __restrict__ Wt,
    const float* __restrict__ bias, float* __restrict__ C,
    int M, int K, int Nout, int act)
{
    __shared__ _Float16 smem[64 * 256];            // 32 KB max (K<=256)
    const int tid  = threadIdx.x;
    const int lane = tid & 31;
    const int wave = tid >> 5;
    const int n0   = blockIdx.y << 6;              // 64-wide N strip

    // cooperative stage of W[n0 .. n0+63][0..K) into LDS (b128 copies)
    const int chunksPerRow = K >> 3;               // v8h chunks per row
    const int totalChunks  = 64 * chunksPerRow;
    for (int idx = tid; idx < totalChunks; idx += 256) {
        int r = idx / chunksPerRow, c = (idx % chunksPerRow) << 3;
        *(v8h*)(smem + r * K + c) = *(const v8h*)(Wt + (size_t)(n0 + r) * K + c);
    }
    __syncthreads();

    const int m0 = (blockIdx.x * 8 + wave) << 4;
    if (m0 >= M) return;                           // wave-uniform (never for our M)

    // A layout (ISA 16-bit A 16x32): lane<16 -> K {0..7, 16..23}; lane>=16 -> +8
    const int kbA = (lane >> 4) << 3;              // 0 or 8
    // B layout (matches A): lane<16 -> K {0..7, 16..23}; lane>=16 -> +8
    const int kbB = (lane >> 4) << 3;              // 0 or 8
    const _Float16* __restrict__ arow = A + (size_t)(m0 + (lane & 15)) * K + kbA;
    const _Float16* srow[4];
#pragma unroll
    for (int j = 0; j < 4; ++j)
        srow[j] = smem + (j * 16 + (lane & 15)) * K + kbB;

    v8f acc[4];
#pragma unroll
    for (int j = 0; j < 4; ++j) acc[j] = (v8f){};

    for (int k0 = 0; k0 < K; k0 += 32) {
        __builtin_prefetch(arow + k0 + 32, 0, 1);
        v8h alo = *(const v8h*)(arow + k0);
        v8h ahi = *(const v8h*)(arow + k0 + 16);
        v16h a = __builtin_shufflevector(alo, ahi, 0, 1, 2, 3, 4, 5, 6, 7,
                                         8, 9, 10, 11, 12, 13, 14, 15);
#pragma unroll
        for (int j = 0; j < 4; ++j) {
            v8h blo = *(const v8h*)(srow[j] + k0);
            v8h bhi = *(const v8h*)(srow[j] + k0 + 16);
            v16h b = __builtin_shufflevector(blo, bhi, 0, 1, 2, 3, 4, 5, 6, 7,
                                             8, 9, 10, 11, 12, 13, 14, 15);
            acc[j] = __builtin_amdgcn_wmma_f32_16x16x32_f16(
                false, a, false, b, (short)0, acc[j], false, false);
        }
    }

    // C/D layout: VGPR i -> M = m0 + i (lanes 0-15) or m0 + 8 + i (lanes 16-31)
    const int mBase = m0 + ((lane >> 4) << 3);
#pragma unroll
    for (int j = 0; j < 4; ++j) {
        const int n    = n0 + j * 16 + (lane & 15);
        const float bv = bias ? bias[n] : 0.0f;
#pragma unroll
        for (int i = 0; i < 8; ++i) {
            float v = acc[j][i] + bv;
            if (act == 1) v = silu_f(v);
            C[(size_t)(mBase + i) * Nout + n] = v;
        }
    }
}

// ---------------------------------------------------------------------------
// Fallback NT GEMM (global B), 16 x 32 strip per wave. Used for Nout == 32.
// ---------------------------------------------------------------------------
__global__ __launch_bounds__(256) void gemm_h16_g2(
    const _Float16* __restrict__ A, const _Float16* __restrict__ Wt,
    const float* __restrict__ bias, float* __restrict__ C,
    int M, int K, int Nout, int act)
{
    const int lane = threadIdx.x & 31;
    const int wave = threadIdx.x >> 5;
    const int nS   = Nout >> 5;
    const int tile = blockIdx.x * 8 + wave;
    if (tile >= (M >> 4) * nS) return;

    const int m0  = (tile / nS) << 4;
    const int n0  = (tile % nS) << 5;
    const int kbA = (lane >> 4) << 3;
    const int kbB = (lane >> 4) << 3;
    const _Float16* __restrict__ arow = A + (size_t)(m0 + (lane & 15)) * K + kbA;
    const _Float16* brow[2];
#pragma unroll
    for (int j = 0; j < 2; ++j)
        brow[j] = Wt + (size_t)(n0 + j * 16 + (lane & 15)) * K + kbB;

    v8f acc[2];
#pragma unroll
    for (int j = 0; j < 2; ++j) acc[j] = (v8f){};

    for (int k0 = 0; k0 < K; k0 += 32) {
        v8h alo = *(const v8h*)(arow + k0);
        v8h ahi = *(const v8h*)(arow + k0 + 16);
        v16h a = __builtin_shufflevector(alo, ahi, 0, 1, 2, 3, 4, 5, 6, 7,
                                         8, 9, 10, 11, 12, 13, 14, 15);
#pragma unroll
        for (int j = 0; j < 2; ++j) {
            v8h blo = *(const v8h*)(brow[j] + k0);
            v8h bhi = *(const v8h*)(brow[j] + k0 + 16);
            v16h b = __builtin_shufflevector(blo, bhi, 0, 1, 2, 3, 4, 5, 6, 7,
                                             8, 9, 10, 11, 12, 13, 14, 15);
            acc[j] = __builtin_amdgcn_wmma_f32_16x16x32_f16(
                false, a, false, b, (short)0, acc[j], false, false);
        }
    }

    const int mBase = m0 + ((lane >> 4) << 3);
#pragma unroll
    for (int j = 0; j < 2; ++j) {
        const int n    = n0 + j * 16 + (lane & 15);
        const float bv = bias ? bias[n] : 0.0f;
#pragma unroll
        for (int i = 0; i < 8; ++i) {
            float v = acc[j][i] + bv;
            if (act == 1) v = silu_f(v);
            C[(size_t)(mBase + i) * Nout + n] = v;
        }
    }
}

// ---------------------------------------------------------------------------
// Elementwise / graph kernels
// ---------------------------------------------------------------------------

__global__ void embed_init_k(const int* __restrict__ z,
                             const float* __restrict__ emb,
                             float* __restrict__ x, float* __restrict__ vec)
{
    int i = blockIdx.x * blockDim.x + threadIdx.x;
    if (i >= 3 * NN * HH) return;
    vec[i] = 0.0f;
    if (i < NN * HH) {
        int node = i >> 7, h = i & 127;
        x[i] = emb[z[node] * HH + h];
    }
}

__global__ void edge_geom_k(const int* __restrict__ ei,
                            const float* __restrict__ pos,
                            const float* __restrict__ betas,
                            const float* __restrict__ means,
                            float* __restrict__ dij, float* __restrict__ cut,
                            float* __restrict__ rbf)
{
    int e = blockIdx.x * blockDim.x + threadIdx.x;
    if (e >= EE) return;
    int s = ei[e], t = ei[EE + e];
    float rx = pos[s * 3 + 0] - pos[t * 3 + 0];
    float ry = pos[s * 3 + 1] - pos[t * 3 + 1];
    float rz = pos[s * 3 + 2] - pos[t * 3 + 2];
    bool loop = (s != t);
    float d2   = rx * rx + ry * ry + rz * rz;
    float dist = loop ? sqrtf(d2) : 0.0f;
    float safe = loop ? dist : 1.0f;
    float inv  = __builtin_amdgcn_rcpf(safe);
    dij[e * 3 + 0] = rx * inv;
    dij[e * 3 + 1] = ry * inv;
    dij[e * 3 + 2] = rz * inv;
    float c = (dist < CUTOFF) ? 0.5f * (cosf(dist * (PI_F / CUTOFF)) + 1.0f) : 0.0f;
    cut[e] = c;
    float ex = __expf(-(5.0f / CUTOFF) * dist);
#pragma unroll
    for (int j = 0; j < NUM_RBF; ++j) {
        float dm = ex - means[j];
        rbf[(size_t)e * NUM_RBF + j] = c * __expf(-betas[j] * dm * dm);
    }
}

__global__ void edge_mul_k(const int* __restrict__ ei,
                           const float* __restrict__ x, float* __restrict__ fij)
{
    int i = blockIdx.x * blockDim.x + threadIdx.x;
    if (i >= EE * HH) return;
    int e = i >> 7, h = i & 127;
    int s = ei[e], t = ei[EE + e];
    fij[i] *= (x[t * HH + h] + x[s * HH + h]);
}

__global__ __launch_bounds__(256) void layernorm_k(
    const float* __restrict__ x, const float* __restrict__ w,
    const float* __restrict__ b, float* __restrict__ y, int rows)
{
    int row  = (blockIdx.x * blockDim.x + threadIdx.x) >> 5;
    int lane = threadIdx.x & 31;
    if (row >= rows) return;
    const float* xr = x + (size_t)row * HH;
    float v0 = xr[lane], v1 = xr[lane + 32], v2 = xr[lane + 64], v3 = xr[lane + 96];
    float s = v0 + v1 + v2 + v3;
    for (int o = 16; o > 0; o >>= 1) s += __shfl_xor(s, o, 32);
    float mu = s * (1.0f / HH);
    float d0 = v0 - mu, d1 = v1 - mu, d2 = v2 - mu, d3 = v3 - mu;
    float q = d0 * d0 + d1 * d1 + d2 * d2 + d3 * d3;
    for (int o = 16; o > 0; o >>= 1) q += __shfl_xor(q, o, 32);
    float inv = rsqrtf(q * (1.0f / HH) + 1e-5f);
    float* yr = y + (size_t)row * HH;
    yr[lane]      = d0 * inv * w[lane]      + b[lane];
    yr[lane + 32] = d1 * inv * w[lane + 32] + b[lane + 32];
    yr[lane + 64] = d2 * inv * w[lane + 64] + b[lane + 64];
    yr[lane + 96] = d3 * inv * w[lane + 96] + b[lane + 96];
}

__global__ void colscale_k(const float* __restrict__ x, const float* __restrict__ w,
                           float* __restrict__ y, int total, int cols)
{
    int i = blockIdx.x * blockDim.x + threadIdx.x;
    if (i >= total) return;
    y[i] = x[i] * w[i % cols];
}

__global__ void vecdot_k(const float* __restrict__ vp, float* __restrict__ vdot)
{
    int i = blockIdx.x * blockDim.x + threadIdx.x;
    if (i >= NN * HH) return;
    int n = i >> 7, h = i & 127;
    float s = 0.0f;
#pragma unroll
    for (int c = 0; c < 3; ++c) {
        size_t r = (size_t)(n * 3 + c) * 384;
        s += vp[r + h] * vp[r + 128 + h];
    }
    vdot[i] = s;
}

__global__ void attn_k(const int* __restrict__ ei, const float* __restrict__ q,
                       const float* __restrict__ k, const float* __restrict__ v,
                       const float* __restrict__ dk, const float* __restrict__ dv,
                       const float* __restrict__ cut, float* __restrict__ vj)
{
    int i = blockIdx.x * blockDim.x + threadIdx.x;
    if (i >= EE * 8) return;
    int e = i >> 3, hd = (i & 7) << 4;
    int s = ei[e], t = ei[EE + e];
    float dot = 0.0f;
#pragma unroll
    for (int d = 0; d < 16; ++d)
        dot += q[t * HH + hd + d] * k[s * HH + hd + d] * dk[(size_t)e * HH + hd + d];
    float a = silu_f(dot) * cut[e];
#pragma unroll
    for (int d = 0; d < 16; ++d)
        vj[(size_t)e * HH + hd + d] = v[s * HH + hd + d] * dv[(size_t)e * HH + hd + d] * a;
}

__global__ void agg_k(const int* __restrict__ ei, const float* __restrict__ vj,
                      const float* __restrict__ svj, const float* __restrict__ vn,
                      const float* __restrict__ dij,
                      float* __restrict__ xagg, float* __restrict__ vagg)
{
    int i = blockIdx.x * blockDim.x + threadIdx.x;
    if (i >= NN * HH) return;
    int n = i >> 7, h = i & 127;
    float xa = 0.0f, va0 = 0.0f, va1 = 0.0f, va2 = 0.0f;
#pragma unroll
    for (int kk = 0; kk < DEG; ++kk) {
        int e = n * DEG + kk;
        int s = ei[e];
        xa += vj[(size_t)e * HH + h];
        float s1 = svj[(size_t)e * 256 + h];
        float s2 = svj[(size_t)e * 256 + 128 + h];
        float dx = dij[e * 3 + 0], dy = dij[e * 3 + 1], dz = dij[e * 3 + 2];
        va0 += vn[(size_t)(s * 3 + 0) * HH + h] * s1 + s2 * dx;
        va1 += vn[(size_t)(s * 3 + 1) * HH + h] * s1 + s2 * dy;
        va2 += vn[(size_t)(s * 3 + 2) * HH + h] * s1 + s2 * dz;
    }
    xagg[i] = xa;
    vagg[(size_t)(n * 3 + 0) * HH + h] = va0;
    vagg[(size_t)(n * 3 + 1) * HH + h] = va1;
    vagg[(size_t)(n * 3 + 2) * HH + h] = va2;
}

__global__ void update_k(const float* __restrict__ oo, const float* __restrict__ vdot,
                         const float* __restrict__ vp, const float* __restrict__ vagg,
                         float* __restrict__ x, float* __restrict__ vec)
{
    int i = blockIdx.x * blockDim.x + threadIdx.x;
    if (i >= NN * HH) return;
    int n = i >> 7, h = i & 127;
    float o1 = oo[(size_t)n * 384 + h];
    float o2 = oo[(size_t)n * 384 + 128 + h];
    float o3 = oo[(size_t)n * 384 + 256 + h];
    x[i] += vdot[i] * o2 + o3;
#pragma unroll
    for (int c = 0; c < 3; ++c) {
        size_t r = (size_t)(n * 3 + c);
        vec[r * HH + h] += vp[r * 384 + 256 + h] * o1 + vagg[r * HH + h];
    }
}

__global__ void fupdate_k(const int* __restrict__ ei, const float* __restrict__ wt,
                          const float* __restrict__ ws, const float* __restrict__ dij,
                          const float* __restrict__ fpj, float* __restrict__ fij)
{
    int i = blockIdx.x * blockDim.x + threadIdx.x;
    if (i >= EE * HH) return;
    int e = i >> 7, h = i & 127;
    int s = ei[e], t = ei[EE + e];
    float dx = dij[e * 3 + 0], dy = dij[e * 3 + 1], dz = dij[e * 3 + 2];
    float a0 = wt[(size_t)(t * 3 + 0) * HH + h];
    float a1 = wt[(size_t)(t * 3 + 1) * HH + h];
    float a2 = wt[(size_t)(t * 3 + 2) * HH + h];
    float b0 = ws[(size_t)(s * 3 + 0) * HH + h];
    float b1 = ws[(size_t)(s * 3 + 1) * HH + h];
    float b2 = ws[(size_t)(s * 3 + 2) * HH + h];
    float vdt = a0 * dx + a1 * dy + a2 * dz;          // w1 . d
    float vds = -(b0 * dx + b1 * dy + b2 * dz);       // w2 . (-d)
    float w10 = a0 - vdt * dx, w11 = a1 - vdt * dy, w12 = a2 - vdt * dz;
    float w20 = b0 + vds * dx, w21 = b1 + vds * dy, w22 = b2 + vds * dz;
    fij[i] += fpj[i] * (w10 * w20 + w11 * w21 + w12 * w22);
}

__global__ void v1norm_k(const float* __restrict__ y, float* __restrict__ v1,
                         int total, int cols)
{
    int i = blockIdx.x * blockDim.x + threadIdx.x;
    if (i >= total) return;
    int n = i / cols, h = i % cols;
    float s = 1e-12f;
#pragma unroll
    for (int c = 0; c < 3; ++c) {
        float t = y[(size_t)(n * 3 + c) * cols + h];
        s += t * t;
    }
    v1[i] = sqrtf(s);
}

__global__ void concat_k(const float* __restrict__ a, const float* __restrict__ b,
                         float* __restrict__ out, int rows, int ca, int cb)
{
    int i = blockIdx.x * blockDim.x + threadIdx.x;
    int cc = ca + cb;
    if (i >= rows * cc) return;
    int r = i / cc, c = i % cc;
    out[i] = (c < ca) ? a[(size_t)r * ca + c] : b[(size_t)r * cb + (c - ca)];
}

__global__ void geb_finish_k(const float* __restrict__ u2, const float* __restrict__ bv2,
                             float* __restrict__ xb, float* __restrict__ vecb,
                             int mid, int uw, int do_silu)
{
    int i = blockIdx.x * blockDim.x + threadIdx.x;
    if (i >= NN * mid) return;
    int n = i / mid, j = i % mid;
    float xv = u2[(size_t)n * uw + j];
    xb[i] = do_silu ? silu_f(xv) : xv;
    float g = u2[(size_t)n * uw + mid + j];
#pragma unroll
    for (int c = 0; c < 3; ++c)
        vecb[(size_t)(n * 3 + c) * mid + j] = g * bv2[(size_t)(n * 3 + c) * mid + j];
}

__global__ void final_k(const float* __restrict__ u2b, float* __restrict__ out)
{
    int i = blockIdx.x * blockDim.x + threadIdx.x;
    if (i >= NN * 32) return;
    int n = i >> 5, j = i & 31;
    out[i] = u2b[(size_t)n * 64 + j];
}

// ---------------------------------------------------------------------------
// Host side
// ---------------------------------------------------------------------------
static inline void cvt(const float* x, _Float16* y, size_t n, hipStream_t stream)
{
    int blocks = (int)((n / 4 + 255) / 256);
    cvt_f16_k<<<blocks, 256, 0, stream>>>(x, y, (int)n);
}

static inline void gemmh(const _Float16* A, const _Float16* W, const float* bias,
                         float* C, int M, int K, int Nout, int act,
                         hipStream_t stream)
{
    if ((Nout & 63) == 0 && (M & 127) == 0) {
        dim3 g(M / 128, Nout / 64);
        gemm_h16_lds<<<g, 256, 0, stream>>>(A, W, bias, C, M, K, Nout, act);
    } else {
        int tiles = (M / 16) * (Nout / 32);
        gemm_h16_g2<<<(tiles + 7) / 8, 256, 0, stream>>>(A, W, bias, C, M, K, Nout, act);
    }
}

#define EW(kern, total, ...) \
    kern<<<(((total) + 255) / 256), 256, 0, stream>>>(__VA_ARGS__)

extern "C" void kernel_launch(void* const* d_in, const int* in_sizes, int n_in,
                              void* d_out, int out_size, void* d_ws, size_t ws_size,
                              hipStream_t stream)
{
    (void)in_sizes; (void)n_in; (void)out_size; (void)ws_size;
    // --- pytree leaves: sorted dict keys, DFS ---------------------------------
    int p = 0;
    auto F = [&](int i) { return (const float*)d_in[i]; };
    const int* batch = (const int*)d_in[p++]; (void)batch;
    const int* eidx  = (const int*)d_in[p++];
    const float *b1u1W = F(p++), *b1u1b = F(p++), *b1u2W = F(p++), *b1u2b = F(p++);
    const float *b1v1W = F(p++), *b1v2W = F(p++);
    const float *b2u1W = F(p++), *b2u1b = F(p++), *b2u2W = F(p++), *b2u2b = F(p++);
    const float *b2v1W = F(p++), *b2v2W = F(p++);
    const float *eeW = F(p++), *eeb = F(p++);
    const float *embedding = F(p++);
    struct Layer {
        const float *dkW, *dkb, *dvW, *dvb, *fW, *fb, *kW, *kb, *lnb, *lnw;
        const float *oW, *ob, *qW, *qb, *sW, *sb, *vW, *vb, *vpW, *vlnw, *wsW, *wtW;
    } L[4];
    for (int li = 0; li < 4; ++li) {
        bool hasF = (li < 3);
        L[li].dkW = F(p++); L[li].dkb = F(p++);
        L[li].dvW = F(p++); L[li].dvb = F(p++);
        if (hasF) { L[li].fW = F(p++); L[li].fb = F(p++); }
        else      { L[li].fW = nullptr; L[li].fb = nullptr; }
        L[li].kW = F(p++); L[li].kb = F(p++);
        L[li].lnb = F(p++); L[li].lnw = F(p++);
        L[li].oW = F(p++); L[li].ob = F(p++);
        L[li].qW = F(p++); L[li].qb = F(p++);
        L[li].sW = F(p++); L[li].sb = F(p++);
        L[li].vW = F(p++); L[li].vb = F(p++);
        L[li].vpW = F(p++); L[li].vlnw = F(p++);
        if (hasF) { L[li].wsW = F(p++); L[li].wtW = F(p++); }
        else      { L[li].wsW = nullptr; L[li].wtW = nullptr; }
    }
    const float *out_ln_b = F(p++), *out_ln_w = F(p++);
    const float *rbf_betas = F(p++), *rbf_means = F(p++);
    const float *vec_out_w = F(p++);
    const float *pos = F(p++);
    const int   *z   = (const int*)d_in[p++];

    // --- fp32 workspace -------------------------------------------------------
    float* base = (float*)d_ws;
    size_t off = 0;
    auto alloc = [&](size_t n) { float* q = base + off; off += n; return q; };
    float* d_dij  = alloc((size_t)EE * 3);
    float* d_cut  = alloc(EE);
    float* d_rbf  = alloc((size_t)EE * NUM_RBF);
    float* d_x    = alloc((size_t)NN * HH);
    float* d_xn   = alloc((size_t)NN * HH);
    float* d_vec  = alloc((size_t)3 * NN * HH);
    float* d_vn   = alloc((size_t)3 * NN * HH);
    float* d_fij  = alloc((size_t)EE * HH);
    float* d_q    = alloc((size_t)NN * HH);
    float* d_k    = alloc((size_t)NN * HH);
    float* d_v    = alloc((size_t)NN * HH);
    float* d_dk   = alloc((size_t)EE * HH);
    float* d_dv   = alloc((size_t)EE * HH);
    float* d_vp   = alloc((size_t)3 * NN * 384);
    float* d_vdot = alloc((size_t)NN * HH);
    float* d_vj   = alloc((size_t)EE * HH);
    float* d_svj  = alloc((size_t)EE * 256);
    float* d_xagg = alloc((size_t)NN * HH);
    float* d_vagg = alloc((size_t)3 * NN * HH);
    float* d_oo   = alloc((size_t)NN * 384);
    float* d_wt   = alloc((size_t)3 * NN * HH);
    float* d_wsp  = alloc((size_t)3 * NN * HH);
    float* d_fpj  = alloc((size_t)EE * HH);
    float* d_cat  = alloc((size_t)NN * 256);
    float* d_u1   = alloc((size_t)NN * HH);
    float* d_u2   = alloc((size_t)NN * HH);
    float* d_bv1  = alloc((size_t)3 * NN * HH);
    float* d_bv2  = alloc((size_t)3 * NN * 64);
    float* d_v1   = alloc((size_t)NN * HH);
    float* d_xb   = alloc((size_t)NN * 64);
    float* d_vecb = alloc((size_t)3 * NN * 64);
    float* d_cv1  = alloc((size_t)3 * NN * 64);
    float* d_cv2  = alloc((size_t)3 * NN * 32);
    float* d_v1b  = alloc((size_t)NN * 64);
    float* d_cat2 = alloc((size_t)NN * HH);
    float* d_u1b  = alloc((size_t)NN * 64);
    float* d_u2b  = alloc((size_t)NN * 64);

    // --- f16 arena (16B aligned: all sizes are multiples of 8) ---------------
    _Float16* hbase = (_Float16*)(base + off);
    size_t hoff = 0;
    auto halloc = [&](size_t n) { _Float16* q = hbase + hoff; hoff += n; return q; };
    _Float16* h_rbf  = halloc((size_t)EE * NUM_RBF);
    _Float16* h_fij  = halloc((size_t)EE * HH);
    _Float16* h_xn   = halloc((size_t)NN * HH);
    _Float16* h_vn   = halloc((size_t)3 * NN * HH);
    _Float16* h_vj   = halloc((size_t)EE * HH);
    _Float16* h_xagg = halloc((size_t)NN * HH);
    _Float16* h_cat  = halloc((size_t)NN * 256);
    _Float16* h_u1   = halloc((size_t)NN * HH);
    _Float16* h_vecb = halloc((size_t)3 * NN * 64);
    _Float16* h_cat2 = halloc((size_t)NN * HH);
    _Float16* h_u1b  = halloc((size_t)NN * 64);
    _Float16* h_eeW  = halloc((size_t)HH * NUM_RBF);
    _Float16 *h_qW[4], *h_kW[4], *h_vW[4], *h_dkW[4], *h_dvW[4];
    _Float16 *h_vpW[4], *h_sW[4], *h_oW[4], *h_fW[4], *h_wsW[4], *h_wtW[4];
    for (int li = 0; li < 4; ++li) {
        h_qW[li]  = halloc((size_t)HH * HH);
        h_kW[li]  = halloc((size_t)HH * HH);
        h_vW[li]  = halloc((size_t)HH * HH);
        h_dkW[li] = halloc((size_t)HH * HH);
        h_dvW[li] = halloc((size_t)HH * HH);
        h_vpW[li] = halloc((size_t)384 * HH);
        h_sW[li]  = halloc((size_t)256 * HH);
        h_oW[li]  = halloc((size_t)384 * HH);
        if (li < 3) {
            h_fW[li]  = halloc((size_t)HH * HH);
            h_wsW[li] = halloc((size_t)HH * HH);
            h_wtW[li] = halloc((size_t)HH * HH);
        } else { h_fW[li] = h_wsW[li] = h_wtW[li] = nullptr; }
    }
    _Float16* h_b1u1W = halloc((size_t)HH * 256);
    _Float16* h_b1u2W = halloc((size_t)HH * HH);
    _Float16* h_b1v1W = halloc((size_t)HH * HH);
    _Float16* h_b1v2W = halloc((size_t)64 * HH);
    _Float16* h_b2u1W = halloc((size_t)64 * HH);
    _Float16* h_b2u2W = halloc((size_t)64 * 64);
    _Float16* h_b2v1W = halloc((size_t)64 * 64);
    _Float16* h_b2v2W = halloc((size_t)32 * 64);

    // --- weight conversion (once per launch) ---------------------------------
    cvt(eeW, h_eeW, (size_t)HH * NUM_RBF, stream);
    for (int li = 0; li < 4; ++li) {
        cvt(L[li].qW,  h_qW[li],  (size_t)HH * HH, stream);
        cvt(L[li].kW,  h_kW[li],  (size_t)HH * HH, stream);
        cvt(L[li].vW,  h_vW[li],  (size_t)HH * HH, stream);
        cvt(L[li].dkW, h_dkW[li], (size_t)HH * HH, stream);
        cvt(L[li].dvW, h_dvW[li], (size_t)HH * HH, stream);
        cvt(L[li].vpW, h_vpW[li], (size_t)384 * HH, stream);
        cvt(L[li].sW,  h_sW[li],  (size_t)256 * HH, stream);
        cvt(L[li].oW,  h_oW[li],  (size_t)384 * HH, stream);
        if (li < 3) {
            cvt(L[li].fW,  h_fW[li],  (size_t)HH * HH, stream);
            cvt(L[li].wsW, h_wsW[li], (size_t)HH * HH, stream);
            cvt(L[li].wtW, h_wtW[li], (size_t)HH * HH, stream);
        }
    }
    cvt(b1u1W, h_b1u1W, (size_t)HH * 256, stream);
    cvt(b1u2W, h_b1u2W, (size_t)HH * HH, stream);
    cvt(b1v1W, h_b1v1W, (size_t)HH * HH, stream);
    cvt(b1v2W, h_b1v2W, (size_t)64 * HH, stream);
    cvt(b2u1W, h_b2u1W, (size_t)64 * HH, stream);
    cvt(b2u2W, h_b2u2W, (size_t)64 * 64, stream);
    cvt(b2v1W, h_b2v1W, (size_t)64 * 64, stream);
    cvt(b2v2W, h_b2v2W, (size_t)32 * 64, stream);

    // --- forward --------------------------------------------------------------
    EW(embed_init_k, 3 * NN * HH, z, embedding, d_x, d_vec);
    EW(edge_geom_k, EE, eidx, pos, rbf_betas, rbf_means, d_dij, d_cut, d_rbf);

    cvt(d_rbf, h_rbf, (size_t)EE * NUM_RBF, stream);
    gemmh(h_rbf, h_eeW, eeb, d_fij, EE, NUM_RBF, HH, 0, stream);
    EW(edge_mul_k, EE * HH, eidx, d_x, d_fij);

    for (int li = 0; li < 4; ++li) {
        const Layer& l = L[li];
        layernorm_k<<<(NN * 32 + 255) / 256, 256, 0, stream>>>(d_x, l.lnw, l.lnb, d_xn, NN);
        EW(colscale_k, 3 * NN * HH, d_vec, l.vlnw, d_vn, 3 * NN * HH, HH);
        cvt(d_xn, h_xn, (size_t)NN * HH, stream);
        cvt(d_vn, h_vn, (size_t)3 * NN * HH, stream);
        cvt(d_fij, h_fij, (size_t)EE * HH, stream);

        gemmh(h_xn, h_qW[li], l.qb, d_q, NN, HH, HH, 0, stream);
        gemmh(h_xn, h_kW[li], l.kb, d_k, NN, HH, HH, 0, stream);
        gemmh(h_xn, h_vW[li], l.vb, d_v, NN, HH, HH, 0, stream);
        gemmh(h_fij, h_dkW[li], l.dkb, d_dk, EE, HH, HH, 1, stream);   // silu
        gemmh(h_fij, h_dvW[li], l.dvb, d_dv, EE, HH, HH, 1, stream);   // silu
        gemmh(h_vn, h_vpW[li], nullptr, d_vp, 3 * NN, HH, 384, 0, stream);

        EW(vecdot_k, NN * HH, d_vp, d_vdot);
        EW(attn_k, EE * 8, eidx, d_q, d_k, d_v, d_dk, d_dv, d_cut, d_vj);
        cvt(d_vj, h_vj, (size_t)EE * HH, stream);
        gemmh(h_vj, h_sW[li], l.sb, d_svj, EE, HH, 256, 1, stream);    // silu
        EW(agg_k, NN * HH, eidx, d_vj, d_svj, d_vn, d_dij, d_xagg, d_vagg);
        cvt(d_xagg, h_xagg, (size_t)NN * HH, stream);
        gemmh(h_xagg, h_oW[li], l.ob, d_oo, NN, HH, 384, 0, stream);
        EW(update_k, NN * HH, d_oo, d_vdot, d_vp, d_vagg, d_x, d_vec);

        if (li < 3) {
            gemmh(h_vn, h_wtW[li], nullptr, d_wt, 3 * NN, HH, HH, 0, stream);
            gemmh(h_vn, h_wsW[li], nullptr, d_wsp, 3 * NN, HH, HH, 0, stream);
            gemmh(h_fij, h_fW[li], l.fb, d_fpj, EE, HH, HH, 1, stream); // silu
            EW(fupdate_k, EE * HH, eidx, d_wt, d_wsp, d_dij, d_fpj, d_fij);
        }
    }

    // output head
    layernorm_k<<<(NN * 32 + 255) / 256, 256, 0, stream>>>(d_x, out_ln_w, out_ln_b, d_xn, NN);
    EW(colscale_k, 3 * NN * HH, d_vec, vec_out_w, d_vn, 3 * NN * HH, HH);
    cvt(d_vn, h_vn, (size_t)3 * NN * HH, stream);

    // GatedEquivariantBlock 1 (scalar_act=True)
    gemmh(h_vn, h_b1v1W, nullptr, d_bv1, 3 * NN, HH, HH, 0, stream);
    EW(v1norm_k, NN * HH, d_bv1, d_v1, NN * HH, HH);
    gemmh(h_vn, h_b1v2W, nullptr, d_bv2, 3 * NN, HH, 64, 0, stream);
    EW(concat_k, NN * 256, d_xn, d_v1, d_cat, NN, HH, HH);
    cvt(d_cat, h_cat, (size_t)NN * 256, stream);
    gemmh(h_cat, h_b1u1W, b1u1b, d_u1, NN, 256, HH, 1, stream);        // silu
    cvt(d_u1, h_u1, (size_t)NN * HH, stream);
    gemmh(h_u1, h_b1u2W, b1u2b, d_u2, NN, HH, HH, 0, stream);
    EW(geb_finish_k, NN * 64, d_u2, d_bv2, d_xb, d_vecb, 64, 128, 1);
    cvt(d_vecb, h_vecb, (size_t)3 * NN * 64, stream);

    // GatedEquivariantBlock 2 (scalar_act=False)
    gemmh(h_vecb, h_b2v1W, nullptr, d_cv1, 3 * NN, 64, 64, 0, stream);
    EW(v1norm_k, NN * 64, d_cv1, d_v1b, NN * 64, 64);
    gemmh(h_vecb, h_b2v2W, nullptr, d_cv2, 3 * NN, 64, 32, 0, stream);
    EW(concat_k, NN * HH, d_xb, d_v1b, d_cat2, NN, 64, 64);
    cvt(d_cat2, h_cat2, (size_t)NN * HH, stream);
    gemmh(h_cat2, h_b2u1W, b2u1b, d_u1b, NN, HH, 64, 1, stream);       // silu
    cvt(d_u1b, h_u1b, (size_t)NN * 64, stream);
    gemmh(h_u1b, h_b2u2W, b2u2b, d_u2b, NN, 64, 64, 0, stream);
    (void)d_cv2; // vector output of block2 is discarded by the reference head

    EW(final_k, NN * 32, d_u2b, (float*)d_out);
}
